// GCN_82952998355125
// MI455X (gfx1250) — matrix operand order repaired and checked by reading, not compiled, not run on hardware
//
#include <hip/hip_runtime.h>

#define NNODES 100000
#define NEDGES 1600000
#define FIN 512
#define FH 64
#define FOUT 20

typedef __attribute__((ext_vector_type(16))) __bf16 v16bf;
typedef __attribute__((ext_vector_type(8)))  float  v8f;

__device__ __forceinline__ unsigned short f32_to_bf16_bits(float f) {
  union { float f; unsigned int u; } c; c.f = f;
  unsigned int u = c.u;
  u += 0x7FFFu + ((u >> 16) & 1u);          // round-to-nearest-even
  return (unsigned short)(u >> 16);
}

__device__ __forceinline__ float selu_f(float x) {
  // scale = 1.0507009873554805, scale*alpha = 1.7580993408473766
  return x > 0.0f ? 1.0507009873554805f * x
                  : 1.7580993408473766f * (__expf(x) - 1.0f);
}

__device__ __forceinline__ v8f wmma_bf16(v16bf a, v16bf b, v8f c) {
  return __builtin_amdgcn_wmma_f32_16x16x32_bf16(false, a, false, b,
                                                 (short)0, c, false, false);
}

// Build a 16x32 bf16 A-fragment for one lane from a row-major f32 row.
// Lane layout (ISA 7.12.2): m = lane&15, g = lane>>4 selects K-half:
//   elems 0..7  -> K = kb .. kb+7        (kb = kt*32 + g*8)
//   elems 8..15 -> K = kb+16 .. kb+23
__device__ __forceinline__ v16bf load_afrag(const float* __restrict__ rowPtr, int kb) {
  const float4 a0 = *(const float4*)(rowPtr + kb);
  const float4 a1 = *(const float4*)(rowPtr + kb + 4);
  const float4 a2 = *(const float4*)(rowPtr + kb + 16);
  const float4 a3 = *(const float4*)(rowPtr + kb + 20);
  v16bf v;
  v[0]  = (__bf16)a0.x; v[1]  = (__bf16)a0.y; v[2]  = (__bf16)a0.z; v[3]  = (__bf16)a0.w;
  v[4]  = (__bf16)a1.x; v[5]  = (__bf16)a1.y; v[6]  = (__bf16)a1.z; v[7]  = (__bf16)a1.w;
  v[8]  = (__bf16)a2.x; v[9]  = (__bf16)a2.y; v[10] = (__bf16)a2.z; v[11] = (__bf16)a2.w;
  v[12] = (__bf16)a3.x; v[13] = (__bf16)a3.y; v[14] = (__bf16)a3.z; v[15] = (__bf16)a3.w;
  return v;
}

// Repack f32 weight [K x Ncols] into fragment-major bf16 B-fragments.
// B layout per lane: n = nt*16 + (lane&15); elem e -> K = kt*32 + (lane>>4)*8 + (e&7) + (e>>3)*16.
// dst index = ((kt*NT + nt)*32 + lane)*16 + e  == linear thread id.
__global__ void pack_weights(const float* __restrict__ src, int Ncols, int NT,
                             unsigned short* __restrict__ dst, int total) {
  int t = blockIdx.x * blockDim.x + threadIdx.x;
  if (t >= total) return;
  int e    = t & 15;
  int lane = (t >> 4) & 31;
  int frag = t >> 9;
  int nt = frag % NT;
  int kt = frag / NT;
  int k = kt * 32 + (lane >> 4) * 8 + (e & 7) + (e >> 3) * 16;
  int n = nt * 16 + (lane & 15);
  float w = (n < Ncols) ? src[k * Ncols + n] : 0.0f;
  dst[t] = f32_to_bf16_bits(w);
}

__global__ void deg_init(float* __restrict__ deg) {
  int i = blockIdx.x * blockDim.x + threadIdx.x;
  if (i < NNODES) deg[i] = 1.0f;                  // self-loop weight
}

__global__ void deg_accum(const int* __restrict__ ei, const float* __restrict__ ew,
                          float* __restrict__ deg) {
  int e = blockIdx.x * blockDim.x + threadIdx.x;
  if (e < NEDGES) atomicAdd(&deg[ei[NEDGES + e]], ew[e]);  // col = ei[1][e]
}

__global__ void deg_to_dinv(float* __restrict__ deg) {
  int i = blockIdx.x * blockDim.x + threadIdx.x;
  if (i < NNODES) {
    float d = deg[i];
    deg[i] = d > 0.0f ? rsqrtf(fmaxf(d, 1e-30f)) : 0.0f;
  }
}

// norm[e] = dinv[row] * w[e] * dinv[col]  (shared by both GCN layers)
__global__ void edge_norm(const int* __restrict__ ei, const float* __restrict__ ew,
                          const float* __restrict__ dinv, float* __restrict__ nrm) {
  int e = blockIdx.x * blockDim.x + threadIdx.x;
  if (e < NEDGES) nrm[e] = dinv[ei[e]] * ew[e] * dinv[ei[NEDGES + e]];
}

// out[N x 64] = A[N x lda] @ Wpacked. 2 M-tiles per wave so each B fragment
// feeds two WMMAs; all 4 B fragments of a k-step loaded before the WMMAs.
template <int KITERS>
__global__ __launch_bounds__(128)
void gemm_bf16(const float* __restrict__ A, int lda,
               const unsigned short* __restrict__ Wp, float* __restrict__ out) {
  const int lane = threadIdx.x & 31;
  const int wave = threadIdx.x >> 5;
  const int g = lane >> 4, j = lane & 15;
  const int rowBase = (blockIdx.x * 4 + wave) * 32;     // 32 rows per wave

  int r0 = rowBase + j;      if (r0 >= NNODES) r0 = NNODES - 1;   // clamp; stores guarded
  int r1 = rowBase + 16 + j; if (r1 >= NNODES) r1 = NNODES - 1;
  const float* Arow0 = A + (long)r0 * lda;
  const float* Arow1 = A + (long)r1 * lda;
  const v16bf* wp = (const v16bf*)Wp;

  v8f acc[2][4] = {};
#pragma unroll
  for (int kt = 0; kt < KITERS; ++kt) {
    v16bf bfr[4];
#pragma unroll
    for (int nt = 0; nt < 4; ++nt) bfr[nt] = wp[kt * 128 + nt * 32 + lane];
    v16bf af0 = load_afrag(Arow0, kt * 32 + g * 8);
    v16bf af1 = load_afrag(Arow1, kt * 32 + g * 8);
#pragma unroll
    for (int nt = 0; nt < 4; ++nt) {
      acc[0][nt] = wmma_bf16(af0, bfr[nt], acc[0][nt]);
      acc[1][nt] = wmma_bf16(af1, bfr[nt], acc[1][nt]);
    }
  }
  // C/D layout: VGPR r -> row = tileBase + g*8 + r, col = nt*16 + j
  if (rowBase + 32 <= NNODES) {                         // uniform fast path
#pragma unroll
    for (int mt = 0; mt < 2; ++mt)
#pragma unroll
      for (int nt = 0; nt < 4; ++nt)
#pragma unroll
        for (int r = 0; r < 8; ++r)
          out[(long)(rowBase + mt * 16 + g * 8 + r) * FH + nt * 16 + j] = acc[mt][nt][r];
  } else {
#pragma unroll
    for (int mt = 0; mt < 2; ++mt)
#pragma unroll
      for (int nt = 0; nt < 4; ++nt)
#pragma unroll
        for (int r = 0; r < 8; ++r) {
          int row = rowBase + mt * 16 + g * 8 + r;
          if (row < NNODES) out[(long)row * FH + nt * 16 + j] = acc[mt][nt][r];
        }
  }
}

// dst[i] = dinv[i]^2 * src[i]  (self-loop contribution, atomic-free init)
__global__ void agg_init(const float* __restrict__ src, const float* __restrict__ dinv,
                         float* __restrict__ dst) {
  int t = blockIdx.x * blockDim.x + threadIdx.x;     // N*16 threads, float4 each
  if (t >= NNODES * 16) return;
  int i = t >> 4, f = (t & 15) * 4;
  float d = dinv[i]; float s = d * d;
  const float4 v = *(const float4*)(src + (long)i * FH + f);
  float4 o; o.x = s * v.x; o.y = s * v.y; o.z = s * v.z; o.w = s * v.w;
  *(float4*)(dst + (long)i * FH + f) = o;
}

// dst[col] += norm[e] * src[row]   (gathers hit L2: table is 25.6MB < 192MB L2)
__global__ void agg_edge(const int* __restrict__ ei, const float* __restrict__ nrm,
                         const float* __restrict__ src, float* __restrict__ dst) {
  int t = blockIdx.x * blockDim.x + threadIdx.x;     // E*16 threads, float4 each
  if (t >= NEDGES * 16) return;
  int e = t >> 4, f = (t & 15) * 4;
  int r = ei[e], c = ei[NEDGES + e];
  float w = nrm[e];
  const float4 v = *(const float4*)(src + (long)r * FH + f);
  float* dp = dst + (long)c * FH + f;
  atomicAdd(dp + 0, w * v.x);
  atomicAdd(dp + 1, w * v.y);
  atomicAdd(dp + 2, w * v.z);
  atomicAdd(dp + 3, w * v.w);
}

__global__ void bias_selu(float* __restrict__ buf, const float* __restrict__ b) {
  int t = blockIdx.x * blockDim.x + threadIdx.x;     // N*16 threads, float4 each
  if (t >= NNODES * 16) return;
  int i = t >> 4, f = (t & 15) * 4;
  float4 v = *(float4*)(buf + (long)i * FH + f);
  v.x = selu_f(v.x + b[f + 0]);
  v.y = selu_f(v.y + b[f + 1]);
  v.z = selu_f(v.z + b[f + 2]);
  v.w = selu_f(v.w + b[f + 3]);
  *(float4*)(buf + (long)i * FH + f) = v;
}

// Fused dense head: 3x (64x64 + SELU) chained through LDS, then 64x20(+pad) + softmax.
__global__ __launch_bounds__(128)
void mlp_head(const float* __restrict__ H,
              const unsigned short* __restrict__ W0p, const float* __restrict__ b0,
              const unsigned short* __restrict__ W1p, const float* __restrict__ b1,
              const unsigned short* __restrict__ W2p, const float* __restrict__ b2,
              const unsigned short* __restrict__ W3p, const float* __restrict__ b3,
              float* __restrict__ outp) {
  __shared__ float lds[4 * 16 * FH];                 // 16KB, per-wave 16x64 tile
  const int lane = threadIdx.x & 31;
  const int wave = threadIdx.x >> 5;
  const int g = lane >> 4, j = lane & 15;
  const int rowBase = (blockIdx.x * 4 + wave) * 16;
  float* my = lds + wave * 16 * FH;

  int arow = rowBase + j; if (arow >= NNODES) arow = NNODES - 1;
  v16bf af0 = load_afrag(H + (long)arow * FH, g * 8);
  v16bf af1 = load_afrag(H + (long)arow * FH, 32 + g * 8);

  const unsigned short* Ws[3] = {W0p, W1p, W2p};
  const float* bs[3] = {b0, b1, b2};
  for (int layer = 0; layer < 3; ++layer) {
    const v16bf* wp = (const v16bf*)Ws[layer];
    const float* bb = bs[layer];
#pragma unroll
    for (int nt = 0; nt < 4; ++nt) {
      v8f a = {};
      a = wmma_bf16(af0, wp[nt * 32 + lane], a);
      a = wmma_bf16(af1, wp[128 + nt * 32 + lane], a);
      int col = nt * 16 + j;
      float bv = bb[col];
#pragma unroll
      for (int r = 0; r < 8; ++r)
        my[(g * 8 + r) * FH + col] = selu_f(a[r] + bv);
    }
    __syncthreads();
    af0 = load_afrag(my + j * FH, g * 8);            // C/D layout -> A layout via LDS
    af1 = load_afrag(my + j * FH, 32 + g * 8);
    __syncthreads();
  }
  {
    const v16bf* wp = (const v16bf*)W3p;             // NT = 2 (20 cols padded to 32)
#pragma unroll
    for (int nt = 0; nt < 2; ++nt) {
      v8f a = {};
      a = wmma_bf16(af0, wp[nt * 32 + lane], a);
      a = wmma_bf16(af1, wp[64 + nt * 32 + lane], a);
      int col = nt * 16 + j;
      float bv = (col < FOUT) ? b3[col] : 0.0f;
#pragma unroll
      for (int r = 0; r < 8; ++r)
        my[(g * 8 + r) * FH + col] = a[r] + bv;
    }
  }
  __syncthreads();
  if (lane < 16) {                                   // one lane per row: softmax over 20
    int row = rowBase + lane;
    if (row < NNODES) {
      const float* rp = my + lane * FH;
      float m = rp[0];
      for (int c = 1; c < FOUT; ++c) m = fmaxf(m, rp[c]);
      float s = 0.0f; float ex[FOUT];
      for (int c = 0; c < FOUT; ++c) { ex[c] = __expf(rp[c] - m); s += ex[c]; }
      float inv = 1.0f / s;
      float* op = outp + (long)row * FOUT;
      for (int c = 0; c < FOUT; ++c) op[c] = ex[c] * inv;
    }
  }
}

extern "C" void kernel_launch(void* const* d_in, const int* in_sizes, int n_in,
                              void* d_out, int out_size, void* d_ws, size_t ws_size,
                              hipStream_t stream) {
  (void)in_sizes; (void)n_in; (void)out_size; (void)ws_size;
  const float* x    = (const float*)d_in[0];
  const int*   ei   = (const int*)d_in[1];
  const float* ew   = (const float*)d_in[2];
  const float* gc0w = (const float*)d_in[3];
  const float* gc0b = (const float*)d_in[4];
  const float* gc1w = (const float*)d_in[5];
  const float* gc1b = (const float*)d_in[6];
  const float* l0w  = (const float*)d_in[7];
  const float* l0b  = (const float*)d_in[8];
  const float* l1w  = (const float*)d_in[9];
  const float* l1b  = (const float*)d_in[10];
  const float* l2w  = (const float*)d_in[11];
  const float* l2b  = (const float*)d_in[12];
  const float* l3w  = (const float*)d_in[13];
  const float* l3b  = (const float*)d_in[14];
  float* outp = (float*)d_out;

  // workspace layout
  float* dinv = (float*)d_ws;                              // N
  float* nrm  = dinv + NNODES;                             // E
  float* bufA = nrm + NEDGES;                              // N*64
  float* bufB = bufA + (size_t)NNODES * FH;                // N*64
  unsigned short* p_gc0 = (unsigned short*)(bufB + (size_t)NNODES * FH); // 32768
  unsigned short* p_gc1 = p_gc0 + 32768;                   // 4096
  unsigned short* p_l0  = p_gc1 + 4096;
  unsigned short* p_l1  = p_l0 + 4096;
  unsigned short* p_l2  = p_l1 + 4096;
  unsigned short* p_l3  = p_l2 + 4096;                     // 2048

  // --- weight repack to fragment-major bf16 ---
  pack_weights<<<32768 / 256, 256, 0, stream>>>(gc0w, 64, 4, p_gc0, 32768);
  pack_weights<<<4096 / 256, 256, 0, stream>>>(gc1w, 64, 4, p_gc1, 4096);
  pack_weights<<<4096 / 256, 256, 0, stream>>>(l0w, 64, 4, p_l0, 4096);
  pack_weights<<<4096 / 256, 256, 0, stream>>>(l1w, 64, 4, p_l1, 4096);
  pack_weights<<<4096 / 256, 256, 0, stream>>>(l2w, 64, 4, p_l2, 4096);
  pack_weights<<<2048 / 256, 256, 0, stream>>>(l3w, FOUT, 2, p_l3, 2048);

  // --- GCN symmetric normalization (computed once, reused by both layers) ---
  deg_init<<<(NNODES + 255) / 256, 256, 0, stream>>>(dinv);
  deg_accum<<<(NEDGES + 255) / 256, 256, 0, stream>>>(ei, ew, dinv);
  deg_to_dinv<<<(NNODES + 255) / 256, 256, 0, stream>>>(dinv);
  edge_norm<<<(NEDGES + 255) / 256, 256, 0, stream>>>(ei, ew, dinv, nrm);

  const int gemmGrid = (NNODES + 127) / 128;               // 32 rows/wave, 4 waves
  const int mlpGrid  = (NNODES + 63) / 64;
  const int elemGrid = (NNODES * 16 + 255) / 256;
  const int edgeGrid = (NEDGES * 16 + 255) / 256;

  // --- GCNConv 0 ---
  gemm_bf16<FIN / 32><<<gemmGrid, 128, 0, stream>>>(x, FIN, p_gc0, bufA);
  agg_init<<<elemGrid, 256, 0, stream>>>(bufA, dinv, bufB);
  agg_edge<<<edgeGrid, 256, 0, stream>>>(ei, nrm, bufA, bufB);
  bias_selu<<<elemGrid, 256, 0, stream>>>(bufB, gc0b);

  // --- GCNConv 1 ---
  gemm_bf16<FH / 32><<<gemmGrid, 128, 0, stream>>>(bufB, FH, p_gc1, bufA);
  agg_init<<<elemGrid, 256, 0, stream>>>(bufA, dinv, bufB);
  agg_edge<<<edgeGrid, 256, 0, stream>>>(ei, nrm, bufA, bufB);
  bias_selu<<<elemGrid, 256, 0, stream>>>(bufB, gc1b);

  // --- fused dense head + softmax ---
  mlp_head<<<mlpGrid, 128, 0, stream>>>(bufB, p_l0, l0b, p_l1, l1b,
                                        p_l2, l2b, p_l3, l3b, outp);
}
